// ScaledDotProductAttention_79748952752779
// MI455X (gfx1250) — matrix operand order repaired and checked by reading, not compiled, not run on hardware
//
#include <hip/hip_runtime.h>

#define S 2048
#define DM 1024

typedef __attribute__((ext_vector_type(16))) __bf16 v16bf;
typedef __attribute__((ext_vector_type(2)))  __bf16 v2bf;
typedef __attribute__((ext_vector_type(8)))  float  v8f;
typedef unsigned short u16;
typedef unsigned int   u32;

union ABv { uint4 u4[2]; v16bf v; };

// native bf16 converts (RNE) -> v_cvt_pk_bf16_f32 on gfx1250
__device__ __forceinline__ u32 pk(float a, float b) {
    v2bf r = { (__bf16)a, (__bf16)b };
    return __builtin_bit_cast(u32, r);
}
__device__ __forceinline__ u16 f2bf(float f) {
    __bf16 h = (__bf16)f;
    return __builtin_bit_cast(u16, h);
}
__device__ __forceinline__ v8f bfwmma(const ABv& a, const ABv& b, v8f c) {
    return __builtin_amdgcn_wmma_f32_16x16x32_bf16(false, a.v, false, b.v,
                                                   (short)0, c, false, false);
}

// fold softmax scale (1/sqrt(64)) and log2(e) into Q so softmax runs on exp2
#define QSCALE 0.18033688011112042f   // 0.125 * 1.4426950408889634

// ---------------- kernel 0: weights -> bf16, transposed [3][64][1024] -------
__global__ void wcvt_kernel(const float* __restrict__ wq,
                            const float* __restrict__ wk,
                            const float* __restrict__ wv,
                            u16* __restrict__ Wt) {
    int t = blockIdx.x * 256 + threadIdx.x;       // 3*64*1024 threads
    if (t >= 3 * 64 * 1024) return;
    int mat = t >> 16;
    int n   = (t >> 10) & 63;
    int k   = t & 1023;
    const float* w = (mat == 0) ? wq : (mat == 1) ? wk : wv;
    Wt[t] = f2bf(w[k * 64 + n]);                  // Wt[mat][n][k] = W[k][n]
}

// ---------------- kernel 1: projections (X @ W) via bf16 WMMA ---------------
// grid (128, 3), block 128.  wave -> one 16-row output tile of one matrix.
__global__ void proj_kernel(const float* __restrict__ x,
                            const u16* __restrict__ Wt,
                            u16* __restrict__ Qs, u16* __restrict__ Ks,
                            u16* __restrict__ Vt) {
    int lane = threadIdx.x & 31, wave = threadIdx.x >> 5;
    int lm = lane & 15, lg = lane >> 4;
    int mat  = blockIdx.y;
    int tile = blockIdx.x * 4 + wave;             // 0..511
    int r0   = tile * 16;
    const float* xrow = x + (size_t)(r0 + lm) * DM;
    const u16*   wt   = Wt + (size_t)mat * 64 * DM;

    v8f acc[4] = {};
    #pragma unroll 2
    for (int kb = 0; kb < DM / 32; ++kb) {
        int k0 = kb * 32 + lg * 8;
        // ---- issue ALL loads first (A: 4 x b128 f32, B: 8 x b128 bf16) ----
        float4 fa0 = *(const float4*)(xrow + k0);
        float4 fa1 = *(const float4*)(xrow + k0 + 4);
        float4 fb0 = *(const float4*)(xrow + k0 + 16);
        float4 fb1 = *(const float4*)(xrow + k0 + 20);
        ABv b[4];
        #pragma unroll
        for (int nt = 0; nt < 4; ++nt) {
            const u16* wr = wt + (size_t)(nt * 16 + lm) * DM + kb * 32 + lg * 8;
            b[nt].u4[0] = *(const uint4*)(wr);
            b[nt].u4[1] = *(const uint4*)(wr + 16);
        }
        ABv a;
        a.u4[0] = make_uint4(pk(fa0.x, fa0.y), pk(fa0.z, fa0.w),
                             pk(fa1.x, fa1.y), pk(fa1.z, fa1.w));
        a.u4[1] = make_uint4(pk(fb0.x, fb0.y), pk(fb0.z, fb0.w),
                             pk(fb1.x, fb1.y), pk(fb1.z, fb1.w));
        #pragma unroll
        for (int nt = 0; nt < 4; ++nt)
            acc[nt] = bfwmma(a, b[nt], acc[nt]);
        // ask the scheduler to interleave next-tile loads between WMMAs
        #pragma unroll
        for (int g = 0; g < 4; ++g) {
            __builtin_amdgcn_sched_group_barrier(0x020, 3, 0); // 3 VMEM read
            __builtin_amdgcn_sched_group_barrier(0x002, 2, 0); // 2 VALU
            __builtin_amdgcn_sched_group_barrier(0x008, 1, 0); // 1 WMMA
        }
    }
    float scale = (mat == 0) ? QSCALE : 1.0f;
    #pragma unroll
    for (int nt = 0; nt < 4; ++nt)
        #pragma unroll
        for (int v = 0; v < 8; ++v) {
            int row = r0 + lg * 8 + v;
            int col = nt * 16 + lm;
            u16 h = f2bf(acc[nt][v] * scale);
            if (mat == 0)      Qs[(size_t)row * 64 + col] = h;
            else if (mat == 1) Ks[(size_t)row * 64 + col] = h;
            else {             // V stored transposed: Vt[b][col][s]
                int b2 = row >> 11, s2 = row & (S - 1);
                Vt[(((size_t)b2 * 64 + col) << 11) + s2] = h;
            }
        }
}

// ---------------- kernel 2: flash attention, one wave per 16 queries --------
__global__ void attn_kernel(const u16* __restrict__ Qs,
                            const u16* __restrict__ Ks,
                            const u16* __restrict__ Vt,
                            const unsigned char* __restrict__ mask,
                            float* __restrict__ out) {
    __shared__ __attribute__((aligned(16))) u16 plds[4][16][32];
    int lane = threadIdx.x & 31, wave = threadIdx.x >> 5;
    int lm = lane & 15, lg = lane >> 4;
    int qt = blockIdx.x * 4 + wave;               // 0..511
    int b  = qt >> 7;
    int q0 = (qt & 127) * 16;

    // Q A-tiles (held for whole loop)
    ABv aq[2];
    #pragma unroll
    for (int ks = 0; ks < 2; ++ks) {
        const u16* qp = Qs + (size_t)(b * S + q0 + lm) * 64 + ks * 32 + lg * 8;
        aq[ks].u4[0] = *(const uint4*)(qp);
        aq[ks].u4[1] = *(const uint4*)(qp + 16);
    }
    // all-ones bf16 B tile: row-sum of P via the matrix pipe (denominator)
    ABv bones;
    bones.u4[0] = make_uint4(0x3F803F80u, 0x3F803F80u, 0x3F803F80u, 0x3F803F80u);
    bones.u4[1] = bones.u4[0];

    const u16* Kb = Ks + (size_t)b * S * 64;
    const u16* Vb = Vt + (size_t)b * 64 * S;
    const unsigned char* mb = mask + (size_t)b * S * S;

    float mrow[8];
    #pragma unroll
    for (int v = 0; v < 8; ++v) mrow[v] = -1e30f;
    v8f acc[4] = {};
    v8f lsum = {};                                // softmax denominator rows

    for (int j = 0; j < S; j += 32) {
        // ---- load ALL K B-tiles for this 32-key block (8 x b128) ----------
        ABv bk[2][2];                             // [keyblk][kstep]
        #pragma unroll
        for (int kblk = 0; kblk < 2; ++kblk)
            #pragma unroll
            for (int ks = 0; ks < 2; ++ks) {
                const u16* kp = Kb + (size_t)(j + kblk * 16 + lm) * 64
                                   + ks * 32 + lg * 8;
                bk[kblk][ks].u4[0] = *(const uint4*)kp;
                bk[kblk][ks].u4[1] = *(const uint4*)(kp + 16);
            }
        // ---- scores: Q[16x64] x K^T[64x32] -> two 16x16 f32 tiles ---------
        v8f sc0 = {}, sc1 = {};
        sc0 = bfwmma(aq[0], bk[0][0], sc0);
        sc0 = bfwmma(aq[1], bk[0][1], sc0);
        sc1 = bfwmma(aq[0], bk[1][0], sc1);
        sc1 = bfwmma(aq[1], bk[1][1], sc1);
        // ---- start V B-tile loads now; latency hides under softmax --------
        ABv bv[4];
        #pragma unroll
        for (int nt = 0; nt < 4; ++nt) {
            const u16* vp = Vb + (size_t)(nt * 16 + lm) * S + j + lg * 8;
            bv[nt].u4[0] = *(const uint4*)vp;
            bv[nt].u4[1] = *(const uint4*)(vp + 16);
        }
        if (j + 32 < S) {
            __builtin_prefetch(Kb + (size_t)(j + 32 + lm) * 64, 0, 1);
            __builtin_prefetch(Vb + (size_t)lm * S + j + 32, 0, 1);
        }
        // ---- mask + online softmax (scores already in log2 domain) --------
        #pragma unroll
        for (int v = 0; v < 8; ++v) {
            int mr = lg * 8 + v;                  // row within tile
            const unsigned char* mp = mb + (size_t)(q0 + mr) * S + j + lm;
            float x0 = mp[0]  ? sc0[v] : -1e30f;
            float x1 = mp[16] ? sc1[v] : -1e30f;
            float rmax = fmaxf(x0, x1);
            #pragma unroll
            for (int off = 1; off < 16; off <<= 1)
                rmax = fmaxf(rmax, __shfl_xor(rmax, off, 32));
            float mnew  = fmaxf(mrow[v], rmax);
            float alpha = exp2f(mrow[v] - mnew);
            mrow[v] = mnew;
            float p0 = exp2f(x0 - mnew);
            float p1 = exp2f(x1 - mnew);
            lsum[v]   *= alpha;                   // denominator rescale
            acc[0][v] *= alpha; acc[1][v] *= alpha;
            acc[2][v] *= alpha; acc[3][v] *= alpha;
            plds[wave][mr][lm]      = f2bf(p0);
            plds[wave][mr][lm + 16] = f2bf(p1);
        }
        __builtin_amdgcn_wave_barrier();          // LDS ops in-order per wave
        // ---- reload P in A-layout -----------------------------------------
        ABv ap;
        const u16* pp = &plds[wave][lm][lg * 8];
        ap.u4[0] = *(const uint4*)(pp);
        ap.u4[1] = *(const uint4*)(pp + 16);
        __builtin_amdgcn_wave_barrier();
        // ---- O += P x V ; lsum += P x ones (row sums on the matrix pipe) --
        #pragma unroll
        for (int nt = 0; nt < 4; ++nt)
            acc[nt] = bfwmma(ap, bv[nt], acc[nt]);
        lsum = bfwmma(ap, bones, lsum);
    }
    // ---- normalize and write fp32 output ----------------------------------
    float* op = out + (size_t)(b * S + q0) * 64;
    #pragma unroll
    for (int v = 0; v < 8; ++v) {
        float inv = 1.0f / lsum[v];
        int rr = lg * 8 + v;
        #pragma unroll
        for (int nt = 0; nt < 4; ++nt)
            op[(size_t)rr * 64 + nt * 16 + lm] = acc[nt][v] * inv;
    }
}

extern "C" void kernel_launch(void* const* d_in, const int* in_sizes, int n_in,
                              void* d_out, int out_size, void* d_ws, size_t ws_size,
                              hipStream_t stream) {
    const unsigned char* mask = (const unsigned char*)d_in[0];
    const float* x  = (const float*)d_in[1];
    const float* wq = (const float*)d_in[2];
    const float* wk = (const float*)d_in[3];
    const float* wv = (const float*)d_in[4];
    float* out = (float*)d_out;

    char* ws = (char*)d_ws;
    u16* Wt = (u16*)(ws);                         // 3*64*1024*2   = 0x060000 B
    u16* Qs = (u16*)(ws + 0x060000);              // 8192*64*2     = 0x100000 B
    u16* Ks = (u16*)(ws + 0x160000);              // 8192*64*2
    u16* Vt = (u16*)(ws + 0x260000);              // 4*64*2048*2   (ends 0x360000)

    wcvt_kernel<<<dim3(768), dim3(256), 0, stream>>>(wq, wk, wv, Wt);
    proj_kernel<<<dim3(128, 3), dim3(128), 0, stream>>>(x, Wt, Qs, Ks, Vt);
    attn_kernel<<<dim3(128), dim3(128), 0, stream>>>(Qs, Ks, Vt, mask, out);
}
